// CouplingLayer_22565758173446
// MI455X (gfx1250) — matrix operand order repaired and checked by reading, not compiled, not run on hardware
//
#include <hip/hip_runtime.h>
#include <hip/hip_fp16.h>

typedef _Float16 v16h __attribute__((ext_vector_type(16)));
typedef _Float16 v8h  __attribute__((ext_vector_type(8)));
typedef float    v8f  __attribute__((ext_vector_type(8)));
typedef float    v2f  __attribute__((ext_vector_type(2)));
typedef int      v4i  __attribute__((vector_size(16)));

#define NB   16
#define HH   64
#define WW   64
#define PIX  (NB*HH*WW)     /* 65536 */
#define HID  128
#define NOUT 272            /* C2 * (C2+1) = 16*17 */
#define NTERMS 18

#if defined(__has_builtin)
#if __has_builtin(__builtin_amdgcn_global_load_async_to_lds_b128)
#define HAVE_ASYNC_LDS 1
#endif
#if __has_builtin(__builtin_amdgcn_s_wait_asynccnt)
#define HAVE_WAIT_ASYNC 1
#endif
#if __has_builtin(__builtin_amdgcn_wmma_f32_16x16x4_f32)
#define HAVE_WMMA_F32K4 1
#endif
#endif

typedef __attribute__((address_space(1))) v4i* as1_v4i;   // global int4*
typedef __attribute__((address_space(3))) v4i* as3_v4i;   // LDS int4*

__device__ __forceinline__ v8f zero8f() {
  v8f r;
#pragma unroll
  for (int i = 0; i < 8; ++i) r[i] = 0.0f;
  return r;
}
__device__ __forceinline__ v8h zero8h() {
  v8h r;
#pragma unroll
  for (int i = 0; i < 8; ++i) r[i] = (_Float16)0.0f;
  return r;
}

// 16-byte global -> LDS copy: async op (ASYNCcnt) when available.
__device__ __forceinline__ void copy16_g2l(const _Float16* g, _Float16* s) {
#if defined(HAVE_ASYNC_LDS)
  __builtin_amdgcn_global_load_async_to_lds_b128(
      (as1_v4i)(unsigned long long)g,                    // 64-bit global addr
      (as3_v4i)(unsigned)(unsigned long long)s,          // 32-bit LDS offset
      0, 0);
#else
  *(v8h*)s = *(const v8h*)g;
#endif
}

template <int N> __device__ __forceinline__ void wait_async() {
#if defined(HAVE_WAIT_ASYNC)
  __builtin_amdgcn_s_wait_asynccnt((short)N);
#elif defined(HAVE_ASYNC_LDS)
  asm volatile("s_wait_asynccnt %0" ::"i"(N) : "memory");
#endif
}

// ---------------------------------------------------------------------------
// Pack input: x (NB,32,64,64) f32 NCHW ->
//   actIn : NHWC f16, C padded to 32 (ch 0..15 = x1, 16..31 = 0)
//   x2buf : [pix][16] f32 ; also copy x1 into d_out channels 0..15
// ---------------------------------------------------------------------------
__global__ __launch_bounds__(256) void pack_x_kernel(
    const float* __restrict__ x, float* __restrict__ out,
    _Float16* __restrict__ actIn, float* __restrict__ x2buf) {
  int idx = blockIdx.x * 256 + threadIdx.x;  // over PIX*32
  if (idx >= PIX * 32) return;
  int c   = idx & 31;
  int pix = idx >> 5;
  int b = pix >> 12, s = pix & 4095;
  float v = x[((size_t)(b * 32 + c)) * 4096 + s];
  if (c < 16) {
    out[((size_t)(b * 32 + c)) * 4096 + s] = v;
    actIn[(size_t)pix * 32 + c] = (_Float16)v;
  } else {
    x2buf[(size_t)pix * 16 + (c - 16)] = v;
    actIn[(size_t)pix * 32 + c] = (_Float16)0.0f;
  }
}

// ---------------------------------------------------------------------------
// Pack conv weights (OIHW f32) into WMMA B-fragment order, f16.
// Two 256-half planes per (kt,nt) fragment so LDS reads are b128 with a
// 16B lane stride (16 lanes span all 64 banks):
//   idx = (kt*NT + nt)*512 + plane*256 + lane*8 + h7 ; hh = plane*8 + h7
//   j = hh + plane*8 + (lane>=16 ? 8 : 0) ; K = kt*32 + j ; g = K/CPAD ; c = K%CPAD
//   n = nt*16 + (lane&15)
// ---------------------------------------------------------------------------
__global__ __launch_bounds__(256) void pack_w_kernel(
    const float* __restrict__ w, _Float16* __restrict__ wp,
    int KT, int NT, int CPAD, int CIN, int COUT) {
  int idx = blockIdx.x * 256 + threadIdx.x;
  int total = KT * NT * 512;
  if (idx >= total) return;
  int h7    = idx & 7;
  int lane  = (idx >> 3) & 31;
  int plane = (idx >> 8) & 1;
  int tmp   = idx >> 9;
  int nt    = tmp % NT;
  int kt    = tmp / NT;
  int hh = plane * 8 + h7;
  int j  = hh + plane * 8 + ((lane & 16) ? 8 : 0);
  int K  = kt * 32 + j;
  int g = K / CPAD, c = K % CPAD;
  int dy = g / 3, dx = g % 3;
  int n = nt * 16 + (lane & 15);
  float v = 0.0f;
  if (c < CIN && n < COUT)
    v = w[(((size_t)n * CIN + c) * 3 + dy) * 3 + dx];
  wp[idx] = (_Float16)v;
}

// ---------------------------------------------------------------------------
// Implicit-GEMM conv, f16 WMMA 16x16x32, f32 accumulate.
// Weights double-buffered in LDS via async global->LDS copies; all 8 waves
// of the block share the staged chunk. A fragments stream from NHWC f16
// activations with predicated zeroing for SAME padding.
// ---------------------------------------------------------------------------
template <int NTC, int KT, int CPAD, int WNT, int KC, bool ELU, bool F32OUT>
__global__ __launch_bounds__(256) void conv_wmma_kernel(
    const _Float16* __restrict__ actIn, const _Float16* __restrict__ wpack,
    const float* __restrict__ bias, _Float16* __restrict__ outH,
    float* __restrict__ outF, int outStride, int ntTot) {
  constexpr int NB128 = KC * WNT * 64;            // 16B blocks per chunk
  constexpr int CLD   = (NB128 + 255) / 256;      // async loads per thread
  constexpr int BUFH  = CLD * 2048;               // halves per buffer (padded)
  constexpr int NC    = KT / KC;                  // chunks
  __shared__ _Float16 sw[2][BUFH];

  const int lane = threadIdx.x & 31;
  const int wave = threadIdx.x >> 5;
  const int mt   = blockIdx.x * 8 + wave;         // M-tile (4096 total)
  const int ntBase = blockIdx.y * NTC;

  const int xb = (mt & 3) * 16;
  const int y  = (mt >> 2) & 63;
  const int b  = mt >> 8;
  const int m  = lane & 15;
  const int hi = (lane >> 4) & 1;
  const int xl = xb + m;

  v8f acc[NTC];
#pragma unroll
  for (int q = 0; q < NTC; ++q) acc[q] = zero8f();

  // issue chunk 0
  {
    const _Float16* wc = wpack;  // chunk 0 base
#pragma unroll
    for (int j = 0; j < CLD; ++j) {
      int i = threadIdx.x + j * 256;
      int isrc = (i < NB128) ? i : (NB128 - 1);
      copy16_g2l(wc + (size_t)isrc * 8, &sw[0][(size_t)i * 8]);
    }
  }

#pragma unroll 1
  for (int c = 0; c < NC; ++c) {
    if (c + 1 < NC) {  // issue next chunk into the other buffer
      const _Float16* wc = wpack + (size_t)(c + 1) * KC * WNT * 512;
#pragma unroll
      for (int j = 0; j < CLD; ++j) {
        int i = threadIdx.x + j * 256;
        int isrc = (i < NB128) ? i : (NB128 - 1);
        copy16_g2l(wc + (size_t)isrc * 8, &sw[(c + 1) & 1][(size_t)i * 8]);
      }
      wait_async<CLD>();  // chunk c complete (async ops retire in order)
    } else {
      wait_async<0>();
    }
    __syncthreads();
    const _Float16* swc = &sw[c & 1][0];

#pragma unroll
    for (int ktl = 0; ktl < KC; ++ktl) {
      const int kt = c * KC + ktl;
      const int g  = (kt * 32) / CPAD;            // CPAD is a power of two
      const int c0 = (kt * 32) % CPAD;
      const int dy = g / 3, dx = g % 3;
      const int ys = y + dy - 1;
      const int xs = xl + dx - 1;
      v8h lo = zero8h(), hv = zero8h();
      if (ys >= 0 && ys < HH && xs >= 0 && xs < WW) {
        const _Float16* ap =
            actIn + ((size_t)(((b * HH) + ys) * WW + xs)) * CPAD + c0 + (hi ? 8 : 0);
        lo = *(const v8h*)ap;         // K c0+{0..7}  (hi lanes: +8..15)
        hv = *(const v8h*)(ap + 16);  // K c0+16+{..} per ISA A16x32 layout
      }
      v16h a;
#pragma unroll
      for (int i = 0; i < 8; ++i) { a[i] = lo[i]; a[i + 8] = hv[i]; }

#pragma unroll
      for (int q = 0; q < NTC; ++q) {
        const int nt = ntBase + q;
        if (nt < ntTot) {                          // wave-uniform branch
          const _Float16* bp = swc + (size_t)(ktl * WNT + nt) * 512 + lane * 8;
          v8h b0 = *(const v8h*)bp;                // plane 0: halves 0..7
          v8h b1 = *(const v8h*)(bp + 256);        // plane 1: halves 8..15
          v16h bb;
#pragma unroll
          for (int i = 0; i < 8; ++i) { bb[i] = b0[i]; bb[i + 8] = b1[i]; }
          acc[q] = __builtin_amdgcn_wmma_f32_16x16x32_f16(
              false, a, false, bb, (short)0, acc[q], false, false);
        }
      }
    }
    __syncthreads();  // all waves done with buffer (c&1) before reuse
  }

  // Epilogue: bias (+ ELU); D layout: vgpr v -> M = v + 8*hi, lane -> N.
#pragma unroll
  for (int q = 0; q < NTC; ++q) {
    const int nt = ntBase + q;
    if (nt < ntTot) {
      const int n = nt * 16 + m;
      const float bv = bias[n];
#pragma unroll
      for (int v = 0; v < 8; ++v) {
        const int mm = v + (hi ? 8 : 0);
        const size_t p2 = (size_t)mt * 16 + mm;
        float val = acc[q][v] + bv;
        if (ELU) val = (val > 0.0f) ? val : expm1f(val);
        if (F32OUT) outF[p2 * outStride + n] = val;
        else        outH[p2 * outStride + n] = (_Float16)val;
      }
    }
  }
}

// ---------------------------------------------------------------------------
// Per-pixel 16x16 expm (18-term Taylor, fully unrolled so 1/k folds) +
// matvec; one wave per pixel. T lives in f32 WMMA B-fragments in registers;
// term round-trips through a wave-private LDS tile (in-order LDS => no
// block barriers, only compiler fences).
// ---------------------------------------------------------------------------
__global__ __launch_bounds__(256) void expm_apply_kernel(
    const float* __restrict__ obuf, const float* __restrict__ x2buf,
    const float* __restrict__ scale_p, const float* __restrict__ shiftp_p,
    const float* __restrict__ rescale_p, const float* __restrict__ reshift_p,
    float* __restrict__ out, float* __restrict__ tracebuf) {
  __shared__ float smem[8 * 544];  // per wave: [0,256) term, [256,272) x2, [272,528) T (fallback)
  const int lane = threadIdx.x & 31;
  const int wave = threadIdx.x >> 5;
  const int pix  = blockIdx.x * 8 + wave;
  float* wm = smem + wave * 544;
  const int n  = lane & 15;
  const int hi = (lane >> 4) & 1;
  const float sc = scale_p[0], sp = shiftp_p[0];
  const float rs = rescale_p[0], rh = reshift_p[0];
  const float* op = obuf + (size_t)pix * NOUT;

  if (lane < 16) wm[256 + lane] = x2buf[(size_t)pix * 16 + lane];

  // T in B-fragment layout: frag kk, elem r: row j = 4kk + 2*hi + r, col n.
  v2f bf[4];
#pragma unroll
  for (int kk = 0; kk < 4; ++kk) {
    int j0 = kk * 4 + (hi ? 2 : 0);
    float r0 = op[(1 + n) * 16 + j0];
    float r1 = op[(1 + n) * 16 + j0 + 1];
    bf[kk][0] = rs * tanhf(sc * r0 + sp) + rh;
    bf[kk][1] = rs * tanhf(sc * r1 + sp) + rh;
#ifndef HAVE_WMMA_F32K4
    wm[272 + (j0)*16 + n]     = bf[kk][0];
    wm[272 + (j0 + 1)*16 + n] = bf[kk][1];
#endif
  }

  // trace(T) -> per-pixel buffer (deterministic tree-reduce later)
  float tr = 0.0f;
  if (lane < 16) {
    float rd = op[(1 + lane) * 16 + lane];
    tr = rs * tanhf(sc * rd + sp) + rh;
  }
#pragma unroll
  for (int off = 16; off >= 1; off >>= 1) tr += __shfl_xor(tr, off, 32);
  if (lane == 0) tracebuf[pix] = tr;

  // term = acc = I   (D layout: vgpr v -> M = v + 8*hi, lane -> N)
  v8f acc;
#pragma unroll
  for (int v = 0; v < 8; ++v) {
    int mm = v + (hi ? 8 : 0);
    float e = (mm == n) ? 1.0f : 0.0f;
    acc[v] = e;
    wm[mm * 16 + n] = e;
  }

#pragma unroll
  for (int k = 1; k <= NTERMS; ++k) {
    asm volatile("" ::: "memory");  // keep LDS store->load order (same wave)
    v8f d = zero8f();
#ifdef HAVE_WMMA_F32K4
    v2f a[4];
#pragma unroll
    for (int kk = 0; kk < 4; ++kk)  // A layout: lane row = n, cols 4kk+2*hi+{0,1}
      a[kk] = *(const v2f*)&wm[n * 16 + kk * 4 + (hi ? 2 : 0)];
    asm volatile("" ::: "memory");
#pragma unroll
    for (int kk = 0; kk < 4; ++kk)
      d = __builtin_amdgcn_wmma_f32_16x16x4_f32(
          false, a[kk], false, bf[kk], (short)0, d, false, false);
#else
#pragma unroll
    for (int v = 0; v < 8; ++v) {
      int mm = v + (hi ? 8 : 0);
      float s2 = 0.0f;
#pragma unroll
      for (int kq = 0; kq < 16; ++kq)
        s2 += wm[mm * 16 + kq] * wm[272 + kq * 16 + n];
      d[v] = s2;
    }
    asm volatile("" ::: "memory");
#endif
    const float inv = 1.0f / (float)k;  // literal after unroll
#pragma unroll
    for (int v = 0; v < 8; ++v) {
      float t = d[v] * inv;
      acc[v] += t;
      wm[(v + (hi ? 8 : 0)) * 16 + n] = t;  // term for next iteration
    }
  }

  asm volatile("" ::: "memory");
#pragma unroll
  for (int v = 0; v < 8; ++v) wm[(v + (hi ? 8 : 0)) * 16 + n] = acc[v];  // E
  asm volatile("" ::: "memory");

  if (lane < 16) {  // y2 = E @ x2 + shift
    float yv = 0.0f;
#pragma unroll
    for (int j = 0; j < 16; ++j) yv += wm[lane * 16 + j] * wm[256 + j];
    yv += op[lane];  // shift = o group 0
    const int b = pix >> 12, s = pix & 4095;
    out[((size_t)(b * 32 + 16 + lane)) * 4096 + s] = yv;
  }
}

// Deterministic log_det: sum 4096 traces per batch with a fixed tree.
__global__ __launch_bounds__(256) void reduce_logdet_kernel(
    const float* __restrict__ tracebuf, float* __restrict__ logdet) {
  __shared__ float s[256];
  int b = blockIdx.x;
  float a = 0.0f;
  for (int i = threadIdx.x; i < 4096; i += 256) a += tracebuf[(size_t)b * 4096 + i];
  s[threadIdx.x] = a;
  __syncthreads();
  for (int st = 128; st > 0; st >>= 1) {
    if (threadIdx.x < st) s[threadIdx.x] += s[threadIdx.x + st];
    __syncthreads();
  }
  if (threadIdx.x == 0) logdet[b] = s[0];
}

// ---------------------------------------------------------------------------
extern "C" void kernel_launch(void* const* d_in, const int* in_sizes, int n_in,
                              void* d_out, int out_size, void* d_ws, size_t ws_size,
                              hipStream_t stream) {
  const float* x       = (const float*)d_in[0];
  const float* scale   = (const float*)d_in[1];
  const float* shift_p = (const float*)d_in[2];
  const float* rescale = (const float*)d_in[3];
  const float* reshift = (const float*)d_in[4];
  const float* w0 = (const float*)d_in[5];  const float* b0 = (const float*)d_in[6];
  const float* w1 = (const float*)d_in[7];  const float* b1 = (const float*)d_in[8];
  const float* w2 = (const float*)d_in[9];  const float* b2 = (const float*)d_in[10];
  const float* w3 = (const float*)d_in[11]; const float* b3 = (const float*)d_in[12];

  float* out    = (float*)d_out;
  float* logdet = out + (size_t)NB * 32 * HH * WW;

  char* ws = (char*)d_ws;
  size_t off = 0;
  auto carve = [&](size_t bytes) { char* p = ws + off; off += (bytes + 255) & ~(size_t)255; return p; };
  _Float16* actIn = (_Float16*)carve((size_t)PIX * 32 * 2);
  _Float16* actA  = (_Float16*)carve((size_t)PIX * HID * 2);
  _Float16* actB  = (_Float16*)carve((size_t)PIX * HID * 2);
  float*    x2buf = (float*)   carve((size_t)PIX * 16 * 4);
  float*    obuf  = (float*)   carve((size_t)PIX * NOUT * 4);
  float*    trbuf = (float*)   carve((size_t)PIX * 4);
  _Float16* w0p   = (_Float16*)carve((size_t)9  * 8  * 512 * 2);
  _Float16* w1p   = (_Float16*)carve((size_t)36 * 8  * 512 * 2);
  _Float16* w2p   = (_Float16*)carve((size_t)36 * 8  * 512 * 2);
  _Float16* w3p   = (_Float16*)carve((size_t)36 * 17 * 512 * 2);
  (void)ws_size; (void)n_in; (void)in_sizes; (void)out_size;

  // stage 0: pack input + copy x1, pack weights into fragment planes
  pack_x_kernel<<<PIX * 32 / 256, 256, 0, stream>>>(x, out, actIn, x2buf);
  pack_w_kernel<<<(9  * 8  * 512 + 255) / 256, 256, 0, stream>>>(w0, w0p, 9,  8,  32,  16,  HID);
  pack_w_kernel<<<(36 * 8  * 512 + 255) / 256, 256, 0, stream>>>(w1, w1p, 36, 8,  HID, HID, HID);
  pack_w_kernel<<<(36 * 8  * 512 + 255) / 256, 256, 0, stream>>>(w2, w2p, 36, 8,  HID, HID, HID);
  pack_w_kernel<<<(36 * 17 * 512 + 255) / 256, 256, 0, stream>>>(w3, w3p, 36, 17, HID, HID, NOUT);

  // stage 1: convs (4096 M-tiles, 8 waves/block, LDS-staged weights)
  conv_wmma_kernel<8, 9, 32, 8, 1, true, false>
      <<<dim3(512, 1), 256, 0, stream>>>(actIn, w0p, b0, actA, nullptr, HID, 8);
  conv_wmma_kernel<8, 36, HID, 8, 2, true, false>
      <<<dim3(512, 1), 256, 0, stream>>>(actA, w1p, b1, actB, nullptr, HID, 8);
  conv_wmma_kernel<8, 36, HID, 8, 2, true, false>
      <<<dim3(512, 1), 256, 0, stream>>>(actB, w2p, b2, actA, nullptr, HID, 8);
  conv_wmma_kernel<6, 36, HID, 17, 1, false, true>
      <<<dim3(512, 3), 256, 0, stream>>>(actA, w3p, b3, nullptr, obuf, NOUT, 17);

  // stage 2: per-pixel expm + apply, then deterministic log_det reduce
  expm_apply_kernel<<<PIX / 8, 256, 0, stream>>>(obuf, x2buf, scale, shift_p,
                                                 rescale, reshift, out, trbuf);
  reduce_logdet_kernel<<<NB, 256, 0, stream>>>(trbuf, logdet);
}
